// GraphMultiHeadAttention_7275674599951
// MI455X (gfx1250) — matrix operand order repaired and checked by reading, not compiled, not run on hardware
//
#include <hip/hip_runtime.h>
#include <hip/hip_bf16.h>

// Problem constants (match reference): N=4096, D=256, H=4, DK=64
#define N_NODES 4096
#define DMODEL  256
#define NHEAD   4
#define DKH     64
#define MASKW   (N_NODES / 32)   // 128 words per row

typedef __attribute__((ext_vector_type(16))) __bf16 v16bf;
typedef __attribute__((ext_vector_type(8)))  __bf16 v8bf;
typedef __attribute__((ext_vector_type(8)))  float  v8f;

union AFrag { v16bf v; struct { v8bf lo, hi; } s; };

__device__ __forceinline__ v8f wmma_bf16(v16bf a, v16bf b, v8f c) {
  // D = A(16x32 bf16) * B(32x16 bf16) + C(16x16 f32)
  return __builtin_amdgcn_wmma_f32_16x16x32_bf16(false, a, false, b, (short)0, c, false, false);
}

// A-matrix 16x32 (MxK): lanes 0-15 row M=lane hold K 0-15,
// lanes 16-31 hold K 16-31. base -> element [M=0][K=0], row stride ld.
__device__ __forceinline__ v16bf load_A(const __bf16* base, int ld, int lane) {
  const int row  = lane & 15;
  const int koff = (lane & 16) ? 16 : 0;
  const __bf16* p = base + row * ld + koff;
  return *(const v16bf*)(p);         // K = koff .. koff+15
}

// B-matrix 32x16 (KxN): col N = lane%16, lanes 0-15 K=0..15, lanes 16-31 K=16..31.
// Stored so that B[k][n] = base[n*ld + k]  (i.e. contiguous in k per lane).
__device__ __forceinline__ v16bf load_B(const __bf16* base, int ld, int lane) {
  const int n  = lane & 15;
  const int kh = (lane & 16) ? 16 : 0;
  return *(const v16bf*)(base + n * ld + kh);
}

// ---------------- mask construction ----------------
__global__ void k_zero_mask(unsigned* mask, int nwords) {
  int i = blockIdx.x * blockDim.x + threadIdx.x;
  if (i < nwords) mask[i] = 0u;
}

__global__ void k_build_mask(const int* ei, int E, unsigned* mask) {
  int e = blockIdx.x * blockDim.x + threadIdx.x;
  if (e >= E) return;
  int src = ei[e];
  int dst = ei[E + e];
  atomicOr(&mask[src * MASKW + (dst >> 5)], 1u << (dst & 31));
}

// ---------------- precision conversion ----------------
__global__ void k_cvt_x(const float* x, __bf16* xbf, int n) {
  int i = blockIdx.x * blockDim.x + threadIdx.x;
  if (i < n) xbf[i] = (__bf16)x[i];
}

// Wt[n][k] = W[k][n] in bf16, for 4 weight matrices (blockIdx.y selects)
__global__ void k_cvt_wt(const float* W0, const float* W1, const float* W2, const float* W3,
                         __bf16* T0, __bf16* T1, __bf16* T2, __bf16* T3) {
  int t = blockIdx.x * blockDim.x + threadIdx.x;          // 0 .. 65535
  const float* W = (blockIdx.y == 0) ? W0 : (blockIdx.y == 1) ? W1 : (blockIdx.y == 2) ? W2 : W3;
  __bf16*      T = (blockIdx.y == 0) ? T0 : (blockIdx.y == 1) ? T1 : (blockIdx.y == 2) ? T2 : T3;
  int k = t >> 8, n = t & 255;
  T[n * DMODEL + k] = (__bf16)W[k * DMODEL + n];
}

// ---------------- QKV projection (WMMA) ----------------
// One wave = one 16x16 output tile. blockIdx.z: 0=Q, 1=K, 2=V(transposed store).
__global__ __launch_bounds__(32) void k_proj(const __bf16* xbf,
                                             const __bf16* Wqt, const __bf16* Wkt, const __bf16* Wvt,
                                             const float* bq, const float* bk, const float* bv,
                                             __bf16* Qbf, __bf16* Kbf, __bf16* Vt) {
  const int lane = threadIdx.x;
  const int row0 = blockIdx.x * 16;
  const int col0 = blockIdx.y * 16;
  const int which = blockIdx.z;
  const __bf16* Wt  = (which == 0) ? Wqt : (which == 1) ? Wkt : Wvt;
  const float*  bia = (which == 0) ? bq  : (which == 1) ? bk  : bv;

  v8f acc = {};
  #pragma unroll
  for (int kt = 0; kt < DMODEL / 32; ++kt) {
    v16bf a = load_A(xbf + (size_t)row0 * DMODEL + kt * 32, DMODEL, lane);
    v16bf b = load_B(Wt  + (size_t)col0 * DMODEL + kt * 32, DMODEL, lane);
    acc = wmma_bf16(a, b, acc);
  }
  const int cl = lane & 15;
  const int hi = (lane & 16) ? 8 : 0;
  const float bb = bia[col0 + cl];
  if (which == 2) {
    // transposed: Vt[d][node], 8 consecutive nodes -> one 16B store
    v8bf pk;
    #pragma unroll
    for (int r = 0; r < 8; ++r) pk[r] = (__bf16)(acc[r] + bb);
    *(v8bf*)(Vt + (size_t)(col0 + cl) * N_NODES + row0 + hi) = pk;
  } else {
    __bf16* out = (which == 0) ? Qbf : Kbf;
    #pragma unroll
    for (int r = 0; r < 8; ++r)
      out[(size_t)(row0 + hi + r) * DMODEL + col0 + cl] = (__bf16)(acc[r] + bb);
  }
}

// ---------------- masked flash attention (WMMA) ----------------
// One wave per (16-query tile, head). Online softmax over 32-wide key steps.
__global__ __launch_bounds__(32) void k_attn(const __bf16* Qbf, const __bf16* Kbf, const __bf16* Vt,
                                             const unsigned* mask, __bf16* ctx) {
  __shared__ __attribute__((aligned(32))) __bf16 Pl[16][32];           // P tile staging (C-layout -> A-layout)
  const int lane = threadIdx.x;
  const int q0 = blockIdx.x * 16;
  const int h  = blockIdx.y;
  const int cl = lane & 15;
  const int hi = (lane & 16) ? 8 : 0;

  const v16bf aQ0 = load_A(Qbf + (size_t)q0 * DMODEL + h * DKH,      DMODEL, lane);
  const v16bf aQ1 = load_A(Qbf + (size_t)q0 * DMODEL + h * DKH + 32, DMODEL, lane);

  v8f acc0 = {}, acc1 = {}, acc2 = {}, acc3 = {};
  float rmax[8], rsum[8];
  #pragma unroll
  for (int r = 0; r < 8; ++r) { rmax[r] = -3.4e38f; rsum[r] = 0.0f; }

  for (int m0 = 0; m0 < N_NODES; m0 += 32) {
    // ---- S = Q K^T for cols [m0, m0+32) : two 16x16 C tiles ----
    v8f z = {};
    v16bf b;
    b = load_B(Kbf + (size_t)m0 * DMODEL + h * DKH,      DMODEL, lane);
    v8f s0 = wmma_bf16(aQ0, b, z);
    b = load_B(Kbf + (size_t)m0 * DMODEL + h * DKH + 32, DMODEL, lane);
    s0 = wmma_bf16(aQ1, b, s0);
    b = load_B(Kbf + (size_t)(m0 + 16) * DMODEL + h * DKH,      DMODEL, lane);
    v8f s1 = wmma_bf16(aQ0, b, z);
    b = load_B(Kbf + (size_t)(m0 + 16) * DMODEL + h * DKH + 32, DMODEL, lane);
    s1 = wmma_bf16(aQ1, b, s1);

    // one mask word covers this wave's 32 key columns for row (q0 + cl)
    unsigned w = mask[(size_t)(q0 + cl) * MASKW + (m0 >> 5)];

    #pragma unroll
    for (int r = 0; r < 8; ++r) {
      unsigned wr = __shfl(w, r + hi);                 // word for this C row
      float v0 = ((wr >> cl)        & 1u) ? s0[r] * 0.125f : -1e9f; // /sqrt(64)
      float v1 = ((wr >> (cl + 16)) & 1u) ? s1[r] * 0.125f : -1e9f;
      // row max across the 16 lanes of this half-wave
      float t = fmaxf(v0, v1);
      t = fmaxf(t, __shfl_xor(t, 1));
      t = fmaxf(t, __shfl_xor(t, 2));
      t = fmaxf(t, __shfl_xor(t, 4));
      t = fmaxf(t, __shfl_xor(t, 8));
      float nm    = fmaxf(rmax[r], t);
      float alpha = __expf(rmax[r] - nm);
      rmax[r] = nm;
      float p0 = __expf(v0 - nm);
      float p1 = __expf(v1 - nm);
      float ps = p0 + p1;
      ps += __shfl_xor(ps, 1);
      ps += __shfl_xor(ps, 2);
      ps += __shfl_xor(ps, 4);
      ps += __shfl_xor(ps, 8);
      rsum[r] = rsum[r] * alpha + ps;
      acc0[r] *= alpha; acc1[r] *= alpha; acc2[r] *= alpha; acc3[r] *= alpha;
      Pl[r + hi][cl]      = (__bf16)p0;   // C-layout -> row-major LDS
      Pl[r + hi][cl + 16] = (__bf16)p1;
    }

    // ---- ctx += P(16x32) @ V(32x64) ----  (LDS reread in A-layout; DS in-order per wave)
    v16bf aP = load_A(&Pl[0][0], 32, lane);
    b = load_B(Vt + (size_t)(h * DKH +  0) * N_NODES + m0, N_NODES, lane);
    acc0 = wmma_bf16(aP, b, acc0);
    b = load_B(Vt + (size_t)(h * DKH + 16) * N_NODES + m0, N_NODES, lane);
    acc1 = wmma_bf16(aP, b, acc1);
    b = load_B(Vt + (size_t)(h * DKH + 32) * N_NODES + m0, N_NODES, lane);
    acc2 = wmma_bf16(aP, b, acc2);
    b = load_B(Vt + (size_t)(h * DKH + 48) * N_NODES + m0, N_NODES, lane);
    acc3 = wmma_bf16(aP, b, acc3);
  }

  #pragma unroll
  for (int r = 0; r < 8; ++r) {
    float inv = 1.0f / rsum[r];
    __bf16* cp = ctx + (size_t)(q0 + hi + r) * DMODEL + h * DKH + cl;
    cp[0]  = (__bf16)(acc0[r] * inv);
    cp[16] = (__bf16)(acc1[r] * inv);
    cp[32] = (__bf16)(acc2[r] * inv);
    cp[48] = (__bf16)(acc3[r] * inv);
  }
}

// ---------------- output projection (WMMA, f32 out) ----------------
__global__ __launch_bounds__(32) void k_outproj(const __bf16* ctx, const __bf16* Wot,
                                                const float* bo, float* out) {
  const int lane = threadIdx.x;
  const int row0 = blockIdx.x * 16;
  const int col0 = blockIdx.y * 16;
  v8f acc = {};
  #pragma unroll
  for (int kt = 0; kt < DMODEL / 32; ++kt) {
    v16bf a = load_A(ctx + (size_t)row0 * DMODEL + kt * 32, DMODEL, lane);
    v16bf b = load_B(Wot + (size_t)col0 * DMODEL + kt * 32, DMODEL, lane);
    acc = wmma_bf16(a, b, acc);
  }
  const int cl = lane & 15;
  const int hi = (lane & 16) ? 8 : 0;
  const float bb = bo[col0 + cl];
  #pragma unroll
  for (int r = 0; r < 8; ++r)
    out[(size_t)(row0 + hi + r) * DMODEL + col0 + cl] = acc[r] + bb;
}

extern "C" void kernel_launch(void* const* d_in, const int* in_sizes, int n_in,
                              void* d_out, int out_size, void* d_ws, size_t ws_size,
                              hipStream_t stream) {
  const float* x  = (const float*)d_in[0];
  const int*   ei = (const int*)d_in[1];
  const float* Wq = (const float*)d_in[2];
  const float* bq = (const float*)d_in[3];
  const float* Wk = (const float*)d_in[4];
  const float* bk = (const float*)d_in[5];
  const float* Wv = (const float*)d_in[6];
  const float* bv = (const float*)d_in[7];
  const float* Wo = (const float*)d_in[8];
  const float* bo = (const float*)d_in[9];
  float* out = (float*)d_out;
  const int E = in_sizes[1] / 2;

  // ---- carve workspace (~12.5 MB total) ----
  char* ws = (char*)d_ws;
  size_t off = 0;
  auto carve = [&](size_t bytes) -> void* {
    void* p = ws + off;
    off += (bytes + 255) & ~(size_t)255;
    return p;
  };
  unsigned* mask = (unsigned*)carve((size_t)N_NODES * MASKW * 4);           // 2 MB
  __bf16* xbf  = (__bf16*)carve((size_t)N_NODES * DMODEL * 2);              // 2 MB
  __bf16* Wqt  = (__bf16*)carve((size_t)DMODEL * DMODEL * 2);
  __bf16* Wkt  = (__bf16*)carve((size_t)DMODEL * DMODEL * 2);
  __bf16* Wvt  = (__bf16*)carve((size_t)DMODEL * DMODEL * 2);
  __bf16* Wot  = (__bf16*)carve((size_t)DMODEL * DMODEL * 2);
  __bf16* Qbf  = (__bf16*)carve((size_t)N_NODES * DMODEL * 2);              // 2 MB
  __bf16* Kbf  = (__bf16*)carve((size_t)N_NODES * DMODEL * 2);              // 2 MB
  __bf16* Vt   = (__bf16*)carve((size_t)DMODEL * N_NODES * 2);              // 2 MB (transposed)
  __bf16* ctxb = (__bf16*)carve((size_t)N_NODES * DMODEL * 2);              // 2 MB

  const int nwords = N_NODES * MASKW;
  k_zero_mask<<<(nwords + 255) / 256, 256, 0, stream>>>(mask, nwords);
  k_build_mask<<<(E + 255) / 256, 256, 0, stream>>>(ei, E, mask);
  k_cvt_x<<<(N_NODES * DMODEL + 255) / 256, 256, 0, stream>>>(x, xbf, N_NODES * DMODEL);
  k_cvt_wt<<<dim3((DMODEL * DMODEL) / 256, 4), 256, 0, stream>>>(Wq, Wk, Wv, Wo, Wqt, Wkt, Wvt, Wot);

  k_proj<<<dim3(N_NODES / 16, DMODEL / 16, 3), 32, 0, stream>>>(
      xbf, Wqt, Wkt, Wvt, bq, bk, bv, Qbf, Kbf, Vt);

  k_attn<<<dim3(N_NODES / 16, NHEAD), 32, 0, stream>>>(Qbf, Kbf, Vt, mask, ctxb);

  k_outproj<<<dim3(N_NODES / 16, DMODEL / 16), 32, 0, stream>>>(ctxb, Wot, bo, out);
}
